// ClassificationPointNetP2_46909632807288
// MI455X (gfx1250) — matrix operand order repaired and checked by reading, not compiled, not run on hardware
//
#include <hip/hip_runtime.h>
#include <hip/hip_bf16.h>
#include <stdint.h>

// ============================================================================
// PointNet++ classification forward for MI455X (gfx1250, wave32, WMMA + TDM).
// GEMMs: v_wmma_f32_16x16x32_f16, A[M,Kp] fp16 row-major, weights transposed
// to Wt[N,Kp] fp16 (one contiguous 32B B-fragment per lane), f32 accumulate,
// fused bias+ReLU. The 16-row A strip for a block is staged into LDS ONCE by
// the Tensor Data Mover (tensor_load_to_lds, TENSORcnt), then the inner loop
// is barrier-free: 2x ds_load_b128 + 2x global_load_b128 + v_wmma per K-step.
// All layer widths padded to multiples of 64 -> no EXEC masking anywhere near
// WMMA. Weights (<3MB fp16) stay resident in the 192MB L2.
// ============================================================================

typedef __attribute__((ext_vector_type(16))) _Float16 v16h;
typedef __attribute__((ext_vector_type(8)))  _Float16 v8h;
typedef __attribute__((ext_vector_type(8)))  float    v8f;
typedef __attribute__((ext_vector_type(4)))  unsigned int v4u;
typedef __attribute__((ext_vector_type(8)))  int      v8i;
typedef __attribute__((ext_vector_type(4)))  int      v4i;

#define NEGV (-1e30f)
#define MAX_KP 1024

// ---------------------------------------------------------------------------
// Generic WMMA GEMM:  C[M,N] = act(A[M,Kp] x W[Kp,N] + bias)
//   block = 128 threads (4 waves); wave w owns cols [by*64 + w*16, +16)
//   N multiple of 64, Kp multiple of 32, M multiple of 16 (all guaranteed).
// ---------------------------------------------------------------------------
__global__ __launch_bounds__(128) void wmma_layer_kernel(
    const _Float16* __restrict__ A, const _Float16* __restrict__ Wt,
    const float* __restrict__ bias, _Float16* __restrict__ C,
    int N, int Kp, int relu)
{
  __shared__ _Float16 sA[16 * MAX_KP];   // whole 16-row A strip (<=32 KB)
  const int t    = threadIdx.x;
  const int lane = t & 31;
  const int w    = t >> 5;
  const int lhi  = lane >> 4;            // 0: lanes 0-15, 1: lanes 16-31
  const int lm   = lane & 15;
  const size_t rowBase = (size_t)blockIdx.x * 16;
  const int colBase = blockIdx.y * 64 + w * 16;

#if __has_builtin(__builtin_amdgcn_tensor_load_to_lds)
  // ---- TDM: one DMA of the [16 x Kp] fp16 tile into LDS (D# per ISA 8.3/8.4)
  if (w == 0) {
    const uint64_t ga = (uint64_t)(const void*)(A + rowBase * (size_t)Kp);
    v4u g0;
    g0[0] = 1u;                                        // count=1, user desc
    g0[1] = 0u;                                        // lds_addr = 0 (sA base)
    g0[2] = (unsigned)ga;                              // global_addr[31:0]
    g0[3] = (unsigned)((ga >> 32) & 0x01FFFFFFu) | (2u << 30);  // addr[56:32]|type=2
    v8i g1;
    g1[0] = 1 << 16;                                   // data_size=1 (2 bytes)
    g1[1] = (Kp & 0xFFFF) << 16;                       // tensor_dim0[15:0]
    g1[2] = ((Kp >> 16) & 0xFFFF) | (16 << 16);        // dim0[31:16] | tensor_dim1=16
    g1[3] = (Kp & 0xFFFF) << 16;                       // dim1[31:16]=0 | tile_dim0=Kp
    g1[4] = 16;                                        // tile_dim1=16, tile_dim2=0
    g1[5] = Kp;                                        // tensor_dim0_stride[31:0]
    g1[6] = 0;                                         // stride[47:32], dim1_stride lo
    g1[7] = 0;
    v4i gz4 = {0, 0, 0, 0};
    v8i gz8 = {0, 0, 0, 0, 0, 0, 0, 0};
    __builtin_amdgcn_tensor_load_to_lds(g0, g1, gz4, gz4, gz8, 0);
    __builtin_amdgcn_s_wait_tensorcnt(0);
  }
  __syncthreads();
#else
  // ---- fallback: cooperative global->LDS staging of the strip
  for (int row = 0; row < 16; ++row)
    for (int col = t * 4; col < Kp; col += 128 * 4)
      *(uint64_t*)(&sA[row * Kp + col]) =
          *(const uint64_t*)(A + (rowBase + row) * (size_t)Kp + col);
  __syncthreads();
#endif
  // The TDM builtin carries no pointer argument, so there is no visible store
  // to sA; escape sA's address into an opaque asm with a memory clobber so
  // the compiler must assume it was written (a bare clobber is NOT enough:
  // escape analysis proves a non-escaping object can't be touched by asm).
  asm volatile("" :: "v"((const void*)&sA[0]) : "memory");

  // A fragment, contiguous k order (same order as B):
  //   lanes 0-15 : m=lane,    e0..15 -> K+0..15
  //   lanes 16-31: m=lane-16, e0..15 -> K+16..31
  // B fragment: lanes 0-15: N=colBase+lm, K+0..15 (contiguous in Wt); hi half K+16..31
  v8f acc = {};
  const _Float16* ap = sA + lm * Kp + lhi * 16;
  const _Float16* wp = Wt + (size_t)(colBase + lm) * Kp + lhi * 16;
  for (int k0 = 0; k0 < Kp; k0 += 32) {
    v8h lo = *(const v8h*)(ap + k0);
    v8h hi = *(const v8h*)(ap + k0 + 8);
    v16h a;
#pragma unroll
    for (int i = 0; i < 8; ++i) { a[i] = lo[i]; a[i + 8] = hi[i]; }
    v16h b = *(const v16h*)(wp + k0);
    acc = __builtin_amdgcn_wmma_f32_16x16x32_f16(false, a, false, b,
                                                 (short)0, acc, false, false);
  }

  const int n = colBase + lm;
  const float bv = bias[n];
#pragma unroll
  for (int i = 0; i < 8; ++i) {          // D layout: VGPR i -> M = lhi*8 + i
    const int m = lhi * 8 + i;
    float v = acc[i] + bv;
    if (relu) v = fmaxf(v, 0.0f);
    C[(rowBase + m) * (size_t)N + n] = (_Float16)v;
  }
}

// ---------------------------------------------------------------------------
// Weight pack: W[K,N] f32  ->  Wt[Np,Kp] fp16 (transposed, zero padded)
// ---------------------------------------------------------------------------
__global__ void pack_wt_kernel(const float* __restrict__ W,
                               _Float16* __restrict__ Wt,
                               int K, int N, int Kp, int total)
{
  int t = blockIdx.x * blockDim.x + threadIdx.x;
  if (t >= total) return;
  int n = t / Kp, k = t % Kp;
  float v = (k < K && n < N) ? W[(size_t)k * N + n] : 0.0f;
  Wt[t] = (_Float16)v;
}

__global__ void padbias_kernel(const float* __restrict__ b2,
                               float* __restrict__ out)
{
  int i = threadIdx.x;             // launched with 64 threads
  out[i] = (i < 2) ? b2[i] : 0.0f;
}

// ---------------------------------------------------------------------------
// Farthest point sampling, one block per cloud (serial argmax chain ->
// latency bound; 16 clouds run concurrently on separate WGPs).
// ---------------------------------------------------------------------------
__global__ void fps_kernel(const float* __restrict__ pos, int Npts, int Mout,
                           int* __restrict__ idx_out, float* __restrict__ q_out)
{
  const int b = blockIdx.x;
  const int tid = threadIdx.x;
  const float* P = pos + (size_t)b * Npts * 3;
  __shared__ float mind[2048];
  __shared__ float rbest[256];
  __shared__ int   rarg[256];
  __shared__ int   s_last;
  __shared__ float lp[3];

  for (int i = tid; i < Npts; i += 256) mind[i] = 1e30f;
  if (tid == 0) {
    s_last = 0;
    idx_out[(size_t)b * Mout] = 0;
    q_out[(size_t)b * Mout * 3 + 0] = P[0];
    q_out[(size_t)b * Mout * 3 + 1] = P[1];
    q_out[(size_t)b * Mout * 3 + 2] = P[2];
  }
  __syncthreads();

  for (int m = 1; m < Mout; ++m) {
    if (tid == 0) {
      int l = s_last;
      lp[0] = P[l * 3 + 0]; lp[1] = P[l * 3 + 1]; lp[2] = P[l * 3 + 2];
    }
    __syncthreads();
    const float lx = lp[0], ly = lp[1], lz = lp[2];
    float bm = -1.0f; int ba = 0x7fffffff;
    for (int i = tid; i < Npts; i += 256) {
      float dx = P[i * 3 + 0] - lx;
      float dy = P[i * 3 + 1] - ly;
      float dz = P[i * 3 + 2] - lz;
      float d = dx * dx + dy * dy + dz * dz;
      float mn = fminf(mind[i], d);
      mind[i] = mn;
      if (mn > bm) { bm = mn; ba = i; }   // strict > keeps lowest index
    }
    rbest[tid] = bm; rarg[tid] = ba;
    __syncthreads();
    for (int s = 128; s > 0; s >>= 1) {
      if (tid < s) {
        float o = rbest[tid + s]; int oa = rarg[tid + s];
        if (o > rbest[tid] || (o == rbest[tid] && oa < rarg[tid])) {
          rbest[tid] = o; rarg[tid] = oa;
        }
      }
      __syncthreads();
    }
    if (tid == 0) {
      int nx = rarg[0];
      s_last = nx;
      idx_out[(size_t)b * Mout + m] = nx;
      q_out[((size_t)b * Mout + m) * 3 + 0] = P[nx * 3 + 0];
      q_out[((size_t)b * Mout + m) * 3 + 1] = P[nx * 3 + 1];
      q_out[((size_t)b * Mout + m) * 3 + 2] = P[nx * 3 + 2];
    }
    __syncthreads();
  }
}

// ---------------------------------------------------------------------------
// Radius top-16: per-query thread, stable insertion into a sorted 16-list
// (matches jax top_k tie ordering: equal scores keep the lower index).
// ---------------------------------------------------------------------------
__global__ void knn_kernel(const float* __restrict__ q,
                           const float* __restrict__ src,
                           int Mq, int Npts, float r2,
                           int* __restrict__ nbr, int* __restrict__ mask,
                           int total)
{
  int t = blockIdx.x * blockDim.x + threadIdx.x;
  if (t >= total) return;
  const int b = t / Mq;
  const float* S = src + (size_t)b * Npts * 3;
  const float qx = q[t * 3 + 0], qy = q[t * 3 + 1], qz = q[t * 3 + 2];

  float sc[16]; int id[16];
#pragma unroll
  for (int i = 0; i < 16; ++i) { sc[i] = -3e38f; id[i] = 0; }

  for (int j = 0; j < Npts; ++j) {
    float dx = S[j * 3 + 0] - qx;
    float dy = S[j * 3 + 1] - qy;
    float dz = S[j * 3 + 2] - qz;
    float d2 = dx * dx + dy * dy + dz * dz;
    float s = (d2 <= r2) ? -d2 : NEGV;
    if (s > sc[15]) {
      sc[15] = s; id[15] = j;
#pragma unroll
      for (int u = 15; u >= 1; --u) {
        if (sc[u] > sc[u - 1]) {
          float ts = sc[u]; sc[u] = sc[u - 1]; sc[u - 1] = ts;
          int   ti = id[u]; id[u] = id[u - 1]; id[u - 1] = ti;
        }
      }
    }
  }
#pragma unroll
  for (int k = 0; k < 16; ++k) {
    nbr [(size_t)t * 16 + k] = id[k];
    mask[(size_t)t * 16 + k] = (sc[k] > NEGV * 0.5f) ? 1 : 0;
  }
}

// ---------------------------------------------------------------------------
// Edge/node feature builders (fp16, zero-padded to Kp)
// ---------------------------------------------------------------------------
__global__ void feat1_kernel(const float* __restrict__ x,
                             const float* __restrict__ pos,
                             const float* __restrict__ p1,
                             const int* __restrict__ nbr,
                             _Float16* __restrict__ out, int total)
{
  int e = blockIdx.x * blockDim.x + threadIdx.x;
  if (e >= total) return;
  const int qi = e >> 4;          // b*1024 + m
  const int b  = qi >> 10;        // M1 = 1024
  const int j  = nbr[e];
  const float* Pb = pos + ((size_t)b * 2048 + j) * 3;
  const float* Xb = x   + ((size_t)b * 2048 + j) * 3;
  const float* Q  = p1  + (size_t)qi * 3;
  _Float16* o = out + (size_t)e * 32;
  o[0] = (_Float16)Xb[0]; o[1] = (_Float16)Xb[1]; o[2] = (_Float16)Xb[2];
  o[3] = (_Float16)(Pb[0] - Q[0]);
  o[4] = (_Float16)(Pb[1] - Q[1]);
  o[5] = (_Float16)(Pb[2] - Q[2]);
#pragma unroll
  for (int c = 6; c < 32; ++c) o[c] = (_Float16)0.0f;
}

__global__ void feat2_kernel(const float* __restrict__ x1,
                             const float* __restrict__ p1,
                             const float* __restrict__ p2,
                             const int* __restrict__ nbr,
                             _Float16* __restrict__ out, int total)
{
  int t = blockIdx.x * blockDim.x + threadIdx.x;
  if (t >= total) return;
  const int e = t / 160, c = t % 160;
  const int qi = e >> 4;          // b*256 + m
  const int b  = qi >> 8;         // M2 = 256
  const int j  = nbr[e];          // index into the M1=1024 nodes of cloud b
  float v;
  if (c < 128)      v = x1[((size_t)b * 1024 + j) * 128 + c];
  else if (c < 131) v = p1[((size_t)b * 1024 + j) * 3 + (c - 128)]
                      - p2[(size_t)qi * 3 + (c - 128)];
  else              v = 0.0f;
  out[(size_t)e * 160 + c] = (_Float16)v;
}

__global__ void feat3_kernel(const float* __restrict__ x2,
                             const float* __restrict__ p2,
                             _Float16* __restrict__ out, int total)
{
  int t = blockIdx.x * blockDim.x + threadIdx.x;
  if (t >= total) return;
  const int node = t / 288, c = t % 288;
  float v;
  if (c < 256)      v = x2[(size_t)node * 256 + c];
  else if (c < 259) v = p2[(size_t)node * 3 + (c - 256)];
  else              v = 0.0f;
  out[(size_t)node * 288 + c] = (_Float16)v;
}

// ---------------------------------------------------------------------------
// Masked max aggregation over K=16 edges per query (scatter-max)
// ---------------------------------------------------------------------------
__global__ void aggmax_kernel(const _Float16* __restrict__ h,
                              const int* __restrict__ mask,
                              float* __restrict__ out, int F)
{
  const int q = blockIdx.x;
  const int c = threadIdx.x;       // blockDim.x == F
  float best = NEGV;
  for (int k = 0; k < 16; ++k) {
    size_t e = (size_t)q * 16 + k;
    if (mask[e]) best = fmaxf(best, (float)h[e * F + c]);
  }
  out[(size_t)q * F + c] = best;
}

// Global max pool over 256 nodes per cloud -> g[B,1024] fp16
__global__ void pool_kernel(const _Float16* __restrict__ h,
                            _Float16* __restrict__ g)
{
  const int b = blockIdx.x;
  for (int c = threadIdx.x; c < 1024; c += blockDim.x) {
    float best = NEGV;
    for (int nd = 0; nd < 256; ++nd)
      best = fmaxf(best, (float)h[((size_t)b * 256 + nd) * 1024 + c]);
    g[(size_t)b * 1024 + c] = (_Float16)best;
  }
}

// Final 2-way softmax from padded [16,64] logits
__global__ void softmax_kernel(const _Float16* __restrict__ logits,
                               float* __restrict__ out)
{
  int b = threadIdx.x;
  if (b >= 16) return;
  float l0 = (float)logits[b * 64 + 0];
  float l1 = (float)logits[b * 64 + 1];
  float mx = fmaxf(l0, l1);
  float e0 = expf(l0 - mx), e1 = expf(l1 - mx);
  float s = e0 + e1;
  out[b * 2 + 0] = e0 / s;
  out[b * 2 + 1] = e1 / s;
}

// ===========================================================================
extern "C" void kernel_launch(void* const* d_in, const int* in_sizes, int n_in,
                              void* d_out, int out_size, void* d_ws, size_t ws_size,
                              hipStream_t stream) {
  (void)in_sizes; (void)n_in; (void)out_size; (void)ws_size;
  const int Bc = 16, M1 = 1024, M2 = 256;
  const int E1 = Bc * M1 * 16;   // 262144 edges
  const int E2 = Bc * M2 * 16;   // 65536 edges
  const int N3 = Bc * M2;        // 4096 nodes

  const float* x_in   = (const float*)d_in[0];
  const float* pos_in = (const float*)d_in[1];
  const float* W[12]  = { (const float*)d_in[2],  (const float*)d_in[4],
                          (const float*)d_in[6],  (const float*)d_in[8],
                          (const float*)d_in[10], (const float*)d_in[12],
                          (const float*)d_in[14], (const float*)d_in[16],
                          (const float*)d_in[18], (const float*)d_in[20],
                          (const float*)d_in[22], (const float*)d_in[24] };
  const float* Bv[12] = { (const float*)d_in[3],  (const float*)d_in[5],
                          (const float*)d_in[7],  (const float*)d_in[9],
                          (const float*)d_in[11], (const float*)d_in[13],
                          (const float*)d_in[15], (const float*)d_in[17],
                          (const float*)d_in[19], (const float*)d_in[21],
                          (const float*)d_in[23], (const float*)d_in[25] };

  // ---- workspace carve-up ----
  char* wsb = (char*)d_ws;
  size_t off = 0;
  auto take = [&](size_t bytes) -> char* {
    off = (off + 255) & ~(size_t)255;
    char* p = wsb + off; off += bytes; return p;
  };
  // packed transposed fp16 weights: dims {K, N, Kp, Np}. Np multiples of 64.
  const int wk[12] = {   6,  64, 128, 131, 128, 128, 259, 256,  512, 1024, 512, 256 };
  const int wn[12] = {  64, 128, 128, 128, 128, 256, 256, 512, 1024,  512, 256,   2 };
  const int wkp[12]= {  32,  64, 128, 160, 128, 128, 288, 256,  512, 1024, 512, 256 };
  const int wnp[12]= {  64, 128, 128, 128, 128, 256, 256, 512, 1024,  512, 256,  64 };
  _Float16* Wt[12];
  for (int i = 0; i < 12; ++i)
    Wt[i] = (_Float16*)take((size_t)wnp[i] * wkp[i] * 2);
  float*    hb2   = (float*)take(64 * 4);
  int*      idx1  = (int*)take((size_t)Bc * M1 * 4);
  float*    p1    = (float*)take((size_t)Bc * M1 * 3 * 4);
  int*      idx2  = (int*)take((size_t)Bc * M2 * 4);
  float*    p2    = (float*)take((size_t)Bc * M2 * 3 * 4);
  int*      nbr1  = (int*)take((size_t)E1 * 4);
  int*      mask1 = (int*)take((size_t)E1 * 4);
  int*      nbr2  = (int*)take((size_t)E2 * 4);
  int*      mask2 = (int*)take((size_t)E2 * 4);
  float*    x1    = (float*)take((size_t)Bc * M1 * 128 * 4);
  float*    x2    = (float*)take((size_t)N3 * 256 * 4);
  _Float16* g     = (_Float16*)take((size_t)Bc * 1024 * 2);
  _Float16* bufA  = (_Float16*)take((size_t)E1 * 128 * 2);   // 64 MiB ping
  _Float16* bufB  = (_Float16*)take((size_t)E1 * 128 * 2);   // 64 MiB pong

  // ---- weight packing (f32 [K,N] -> fp16 transposed [Np,Kp]) ----
  for (int i = 0; i < 12; ++i) {
    int total = wnp[i] * wkp[i];
    pack_wt_kernel<<<dim3((total + 255) / 256), dim3(256), 0, stream>>>(
        W[i], Wt[i], wk[i], wn[i], wkp[i], total);
  }
  padbias_kernel<<<dim3(1), dim3(64), 0, stream>>>(Bv[11], hb2);

  auto gemm = [&](const _Float16* A, int wi, _Float16* C,
                  int M, const float* bias, int relu) {
    dim3 gg(M / 16, wnp[wi] / 64);
    wmma_layer_kernel<<<gg, dim3(128), 0, stream>>>(
        A, Wt[wi], bias, C, wnp[wi], wkp[wi], relu);
  };

  // ---- SA1: FPS(2048->1024), radius 0.2, edge MLP 6->64->128->128, max ----
  fps_kernel<<<dim3(Bc), dim3(256), 0, stream>>>(pos_in, 2048, M1, idx1, p1);
  knn_kernel<<<dim3((Bc * M1 + 127) / 128), dim3(128), 0, stream>>>(
      p1, pos_in, M1, 2048, 0.04f, nbr1, mask1, Bc * M1);
  feat1_kernel<<<dim3((E1 + 127) / 128), dim3(128), 0, stream>>>(
      x_in, pos_in, p1, nbr1, bufA, E1);
  gemm(bufA, 0, bufB, E1, Bv[0], 1);     // [E1,32]  x [32,64]
  gemm(bufB, 1, bufA, E1, Bv[1], 1);     // [E1,64]  x [64,128]
  gemm(bufA, 2, bufB, E1, Bv[2], 0);     // [E1,128] x [128,128]
  aggmax_kernel<<<dim3(Bc * M1), dim3(128), 0, stream>>>(bufB, mask1, x1, 128);

  // ---- SA2: FPS(1024->256), radius 0.4, edge MLP 131->128->128->256, max --
  fps_kernel<<<dim3(Bc), dim3(256), 0, stream>>>(p1, M1, M2, idx2, p2);
  knn_kernel<<<dim3((Bc * M2 + 127) / 128), dim3(128), 0, stream>>>(
      p2, p1, M2, M1, 0.16f, nbr2, mask2, Bc * M2);
  feat2_kernel<<<dim3((E2 * 160 + 255) / 256), dim3(256), 0, stream>>>(
      x1, p1, p2, nbr2, bufA, E2 * 160);
  gemm(bufA, 3, bufB, E2, Bv[3], 1);     // [E2,160] x [160,128]
  gemm(bufB, 4, bufA, E2, Bv[4], 1);     // [E2,128] x [128,128]
  gemm(bufA, 5, bufB, E2, Bv[5], 0);     // [E2,128] x [128,256]
  aggmax_kernel<<<dim3(Bc * M2), dim3(256), 0, stream>>>(bufB, mask2, x2, 256);

  // ---- SA3 (global): node MLP 259->256->512->1024, global max pool -------
  feat3_kernel<<<dim3((N3 * 288 + 255) / 256), dim3(256), 0, stream>>>(
      x2, p2, bufA, N3 * 288);
  gemm(bufA, 6, bufB, N3, Bv[6], 1);     // [4096,288] x [288,256]
  gemm(bufB, 7, bufA, N3, Bv[7], 1);     // [4096,256] x [256,512]
  gemm(bufA, 8, bufB, N3, Bv[8], 0);     // [4096,512] x [512,1024]
  pool_kernel<<<dim3(Bc), dim3(256), 0, stream>>>(bufB, g);

  // ---- head: 1024->512->256->2 (N padded to 64), softmax ------------------
  gemm(g,    9,  bufA, 16, Bv[9],  1);   // [16,1024] x [1024,512]
  gemm(bufA, 10, bufB, 16, Bv[10], 1);   // [16,512]  x [512,256]
  gemm(bufB, 11, bufA, 16, hb2,    0);   // [16,256]  x [256,64]
  softmax_kernel<<<dim3(1), dim3(32), 0, stream>>>(bufA, (float*)d_out);
}
